// TransformerEncoderLayer_81518479278705
// MI455X (gfx1250) — matrix operand order repaired and checked by reading, not compiled, not run on hardware
//
#include <hip/hip_runtime.h>
#include <cstdint>
#include <cstddef>

// ---------- types ----------
typedef unsigned short u16;
typedef __bf16 bf16_t;
typedef bf16_t v16bf __attribute__((ext_vector_type(16)));
typedef float  v8f   __attribute__((ext_vector_type(8)));

struct alignas(16) U128 { unsigned x, y, z, w; };
struct F32x8 { U128 a, b; };   // 32 bytes == one v16bf fragment

__device__ __forceinline__ v16bf as_frag(const U128& lo, const U128& hi) {
    F32x8 t{lo, hi};
    return __builtin_bit_cast(v16bf, t);
}

__device__ __forceinline__ u16 f2bf(float f) {  // round-to-nearest-even
    union { float f; unsigned u; } v; v.f = f;
    unsigned r = v.u + 0x7FFFu + ((v.u >> 16) & 1u);
    return (u16)(r >> 16);
}

__device__ __forceinline__ v8f wmma_bf16(v16bf a, v16bf b, v8f c) {
    return __builtin_amdgcn_wmma_f32_16x16x32_bf16(false, a, false, b, (short)0, c, false, false);
}

// ---------- problem constants ----------
#define BB   2
#define SS   2048
#define DD   1024
#define HH   16
#define DHH  64
#define FFNN 4096
#define MM   (BB * SS)         // 4096 token rows
#define TD   (3 * DD)          // 3072

// LDS row strides (elements), padded so the 16 lanes of a ds_load_b128
// phase spread over all 64 banks; strides stay 16-byte aligned.
#define TSTR 72                // 128x64 tiles: 144 B / row
#define VSTR 40                // 32-key tiles: 80 B / row

// =====================================================================
// async global -> LDS tile loader (CDNA5 GLOBAL_LOAD_ASYNC_TO_LDS_B128,
// GVS addressing: mem = SGPR64 + VGPR32, tracked by ASYNCcnt).
// Loads a 128 x 64 bf16 tile (row stride K in global, TSTR in LDS)
// with NT threads: 1024 chunks of 16 B.
// =====================================================================
template <int NT>
__device__ __forceinline__ void async_tile_load(
    const u16* __restrict__ gbase, int K, int k0, u16* lds, int tid) {
#pragma unroll
    for (int i = 0; i < 1024 / NT; ++i) {
        const int idx = tid + i * NT;
        const int r = idx >> 3, ck = idx & 7;
        const unsigned voff  = (unsigned)((r * K + k0 + ck * 8) * 2);
        const unsigned laddr = (unsigned)(uintptr_t)(lds + r * TSTR + ck * 8);
        asm volatile("global_load_async_to_lds_b128 %0, %1, %2"
                     :: "v"(laddr), "v"(voff), "s"(gbase)
                     : "memory");
    }
}

__device__ __forceinline__ void wait_async0() {
    asm volatile("s_wait_asynccnt 0x0" ::: "memory");
}

// =====================================================================
// Kernel 1: fp32 [R][C] -> bf16 [C][R]  (weights into B-operand layout)
// =====================================================================
__global__ __launch_bounds__(256) void k_transpose_bf16(
    const float* __restrict__ in, u16* __restrict__ out, int R, int C) {
    __shared__ float tile[32][33];
    const int c0 = blockIdx.x * 32, r0 = blockIdx.y * 32;
    const int tid = threadIdx.x;
#pragma unroll
    for (int i = 0; i < 4; ++i) {
        int idx = tid + i * 256, r = idx >> 5, c = idx & 31;
        tile[r][c] = in[(size_t)(r0 + r) * C + (c0 + c)];
    }
    __syncthreads();
#pragma unroll
    for (int i = 0; i < 4; ++i) {
        int idx = tid + i * 256, r = idx >> 5, c = idx & 31;   // r = local n, c = local k
        out[(size_t)(c0 + r) * R + (r0 + c)] = f2bf(tile[c][r]);
    }
}

// =====================================================================
// Kernel 2: row LayerNorm (fp32 in) -> bf16 out.  One block per row.
// =====================================================================
__global__ __launch_bounds__(256) void k_layernorm_bf16(
    const float* __restrict__ x, const float* __restrict__ g,
    const float* __restrict__ be, u16* __restrict__ out, int D) {
    const int row = blockIdx.x, tid = threadIdx.x;
    const float* xr = x + (size_t)row * D;
    float s = 0.f, s2 = 0.f;
    for (int i = tid; i < D; i += 256) { float v = xr[i]; s += v; s2 += v * v; }
    __shared__ float red[2][256];
    red[0][tid] = s; red[1][tid] = s2;
    __syncthreads();
    for (int off = 128; off > 0; off >>= 1) {
        if (tid < off) { red[0][tid] += red[0][tid + off]; red[1][tid] += red[1][tid + off]; }
        __syncthreads();
    }
    const float mu  = red[0][0] / (float)D;
    const float var = red[1][0] / (float)D - mu * mu;
    const float rs  = rsqrtf(var + 1e-5f);
    for (int i = tid; i < D; i += 256)
        out[(size_t)row * D + i] = f2bf((xr[i] - mu) * rs * g[i] + be[i]);
}

// =====================================================================
// Kernel 3: bf16 GEMM  C[M,N] = A[M,K] @ Bt[N,K]^T  (+bias)(+relu)(+resid)
// Block tile 128x128, K-step 64, double-buffered LDS fed by async-to-LDS,
// one barrier per K-step.  4 waves (2x2); wave tile 64x64 = 4x4 WMMA:
// 16 WMMA per 8 fragment loads per 32-deep sub-slice.
// =====================================================================
template <bool BIAS, bool RELU, bool RESID, bool OUTBF>
__global__ __launch_bounds__(128) void k_gemm_bf16(
    const u16* __restrict__ A, const u16* __restrict__ Bt,
    const float* __restrict__ bias, const float* __restrict__ resid,
    void* __restrict__ outv, int M, int N, int K) {
    __shared__ u16 As[2][128 * TSTR];   // 2 x 18 KB
    __shared__ u16 Bs[2][128 * TSTR];   // 2 x 18 KB
    const int tid   = threadIdx.x;
    const int lane  = tid & 31, wid = tid >> 5;
    const int l15   = lane & 15, lhalf = lane >> 4;
    const int rowB  = blockIdx.y * 128, colB = blockIdx.x * 128;
    const int wm    = (wid & 1) * 64, wn = (wid >> 1) * 64;

    const u16* agp = A  + (size_t)rowB * K;   // uniform block bases (SGPRs)
    const u16* bgp = Bt + (size_t)colB * K;

    v8f acc[4][4];
#pragma unroll
    for (int mi = 0; mi < 4; ++mi)
#pragma unroll
        for (int ni = 0; ni < 4; ++ni)
            acc[mi][ni] = v8f{0.f, 0.f, 0.f, 0.f, 0.f, 0.f, 0.f, 0.f};

    // prologue: stage 0 in flight
    async_tile_load<128>(agp, K, 0, &As[0][0], tid);
    async_tile_load<128>(bgp, K, 0, &Bs[0][0], tid);

    int p = 0;
    for (int k0 = 0; k0 < K; k0 += 64) {
        wait_async0();          // stage p landed in LDS (this wave's share)
        __syncthreads();        // all waves' shares landed; buf p^1 free
        if (k0 + 64 < K) {      // launch next stage into the other buffer
            async_tile_load<128>(agp, K, k0 + 64, &As[p ^ 1][0], tid);
            async_tile_load<128>(bgp, K, k0 + 64, &Bs[p ^ 1][0], tid);
        }
        const u16* as = &As[p][0];
        const u16* bs = &Bs[p][0];
#pragma unroll
        for (int ks = 0; ks < 2; ++ks) {      // two 32-deep WMMA sub-slices
            v16bf af[4], bf[4];
#pragma unroll
            for (int mi = 0; mi < 4; ++mi) {  // A frag: k = {8h..}, {16+8h..}
                int r = wm + mi * 16 + l15;
                af[mi] = as_frag(
                    *(const U128*)(as + r * TSTR + ks * 32 + 8 * lhalf),
                    *(const U128*)(as + r * TSTR + ks * 32 + 16 + 8 * lhalf));
            }
#pragma unroll
            for (int ni = 0; ni < 4; ++ni) {  // B frag: k = {8h..}, {16+8h..}
                int n = wn + ni * 16 + l15;
                bf[ni] = as_frag(
                    *(const U128*)(bs + n * TSTR + ks * 32 + 8 * lhalf),
                    *(const U128*)(bs + n * TSTR + ks * 32 + 16 + 8 * lhalf));
            }
#pragma unroll
            for (int mi = 0; mi < 4; ++mi)
#pragma unroll
                for (int ni = 0; ni < 4; ++ni)
                    acc[mi][ni] = wmma_bf16(af[mi], bf[ni], acc[mi][ni]);
        }
        p ^= 1;
    }

    // epilogue: C layout row = j + 8*lhalf, col = l15
#pragma unroll
    for (int ni = 0; ni < 4; ++ni) {
        const int col = colB + wn + ni * 16 + l15;
        float bv = 0.f;
        if constexpr (BIAS) bv = bias[col];
#pragma unroll
        for (int mi = 0; mi < 4; ++mi) {
#pragma unroll
            for (int j = 0; j < 8; ++j) {
                const int row = rowB + wm + mi * 16 + j + 8 * lhalf;
                float v = acc[mi][ni][j] + bv;
                if constexpr (RELU) v = fmaxf(v, 0.f);
                const size_t oi = (size_t)row * N + col;
                if constexpr (RESID) v += resid[oi];
                if constexpr (OUTBF) ((u16*)outv)[oi] = f2bf(v);
                else                 ((float*)outv)[oi] = v;
            }
        }
    }
}

// =====================================================================
// Kernel 4: flash attention.  Block = (b, h, 128 q-rows); 8 waves, each
// wave owns 16 q rows (spill-free register budget: 32 acc + 16 Q-frag +
// 16 softmax-state VGPRs).  Key tile = 32, register-prefetched one tile
// ahead (global latency hidden behind WMMA + softmax of current tile);
// K/V re-reads across q-blocks hit the 192 MB L2 (whole qkv is 25 MB).
// qkv bf16 layout: [B*S][3*D], Q at col h*64, K at D+h*64, V at 2D+h*64.
// =====================================================================
__global__ __launch_bounds__(256) void k_attention(
    const u16* __restrict__ qkv, const int* __restrict__ mask,
    u16* __restrict__ out) {
    __shared__ float maskadd[SS];            // 8 KB
    __shared__ u16 Kt[32 * TSTR];            // [key][dh]  padded
    __shared__ u16 Vt[64 * VSTR];            // [dh][key]  padded (transposed V)
    __shared__ u16 Pt[8][16 * VSTR];         // per-wave P, padded

    const int tid = threadIdx.x, lane = tid & 31, wid = tid >> 5;
    const int l15 = lane & 15, lhalf = lane >> 4;
    const int qb = blockIdx.x * 128, h = blockIdx.y, b = blockIdx.z;
    const float scale = 0.125f;              // 1/sqrt(64)

    for (int i = tid; i < SS; i += 256)
        maskadd[i] = mask[b * SS + i] ? 0.f : -1e9f;

    // Q fragments (A layout), loaded straight from global
    const int qrow = qb + wid * 16 + l15;
    const u16* qp = qkv + (size_t)(b * SS + qrow) * TD + h * DHH;
    v16bf qf[2];
#pragma unroll
    for (int kh = 0; kh < 2; ++kh)
        qf[kh] = as_frag(*(const U128*)(qp + kh * 32 + 8 * lhalf),
                         *(const U128*)(qp + kh * 32 + 16 + 8 * lhalf));

    v8f acc[4];
#pragma unroll
    for (int dt = 0; dt < 4; ++dt) acc[dt] = v8f{0.f,0.f,0.f,0.f,0.f,0.f,0.f,0.f};
    float mrow[8], lrow[8];
#pragma unroll
    for (int j = 0; j < 8; ++j) { mrow[j] = -1e30f; lrow[j] = 0.f; }

    // register-prefetch pipeline for K/V tiles
    const int key = tid >> 3, ck = tid & 7;
    U128 kreg, vreg;
    {
        const u16* kr = qkv + (size_t)(b * SS + key) * TD + h * DHH;
        kreg = *(const U128*)(kr + DD + ck * 8);
        vreg = *(const U128*)(kr + 2 * DD + ck * 8);
    }

    for (int kt = 0; kt < SS; kt += 32) {
        __syncthreads();                       // everyone done reading prev tile
        *(U128*)(Kt + key * TSTR + ck * 8) = kreg;
        {
            const u16* pv = (const u16*)&vreg; // scatter-transpose V into LDS
#pragma unroll
            for (int e = 0; e < 8; ++e)
                Vt[(ck * 8 + e) * VSTR + key] = pv[e];
        }
        __syncthreads();
        if (kt + 32 < SS) {                    // prefetch next tile (consumed next iter)
            const u16* kr = qkv + (size_t)(b * SS + kt + 32 + key) * TD + h * DHH;
            kreg = *(const U128*)(kr + DD + ck * 8);
            vreg = *(const U128*)(kr + 2 * DD + ck * 8);
        }

        // scores S = Q @ K^T  (16 x 32)
        v8f sc[2];
#pragma unroll
        for (int nt = 0; nt < 2; ++nt) {
            v8f c = v8f{0.f,0.f,0.f,0.f,0.f,0.f,0.f,0.f};
#pragma unroll
            for (int kh = 0; kh < 2; ++kh) {
                const int n = nt * 16 + l15;   // key col; k = dh, contiguous per key row
                v16bf bfr = as_frag(
                    *(const U128*)(Kt + n * TSTR + kh * 32 + 8 * lhalf),
                    *(const U128*)(Kt + n * TSTR + kh * 32 + 16 + 8 * lhalf));
                c = wmma_bf16(qf[kh], bfr, c);
            }
            sc[nt] = c;
        }

        const float ma0 = maskadd[kt + l15];
        const float ma1 = maskadd[kt + 16 + l15];

        // online softmax per row (rows j + 8*lhalf; 16-lane reductions)
#pragma unroll
        for (int j = 0; j < 8; ++j) {
            float s0 = sc[0][j] * scale + ma0;
            float s1 = sc[1][j] * scale + ma1;
            float rm = fmaxf(s0, s1);
#pragma unroll
            for (int off = 1; off < 16; off <<= 1)
                rm = fmaxf(rm, __shfl_xor(rm, off, 32));
            const float nm = fmaxf(mrow[j], rm);
            const float p0 = __expf(s0 - nm), p1 = __expf(s1 - nm);
            float rs = p0 + p1;
#pragma unroll
            for (int off = 1; off < 16; off <<= 1)
                rs += __shfl_xor(rs, off, 32);
            const float cf = __expf(mrow[j] - nm);
            mrow[j] = nm;
            lrow[j] = lrow[j] * cf + rs;
#pragma unroll
            for (int dt = 0; dt < 4; ++dt) acc[dt] *= cf;
            const int prow = j + 8 * lhalf;
            Pt[wid][prow * VSTR + l15]      = f2bf(p0);
            Pt[wid][prow * VSTR + 16 + l15] = f2bf(p1);
        }
        // in-wave LDS transpose: wait for our own ds stores before reloading
        asm volatile("s_wait_dscnt 0x0" ::: "memory");

        // P (16x32) as A fragment, V^T columns as B fragments
        v16bf pf = as_frag(*(const U128*)(&Pt[wid][l15 * VSTR + 8 * lhalf]),
                           *(const U128*)(&Pt[wid][l15 * VSTR + 16 + 8 * lhalf]));
#pragma unroll
        for (int dt = 0; dt < 4; ++dt) {
            const int n = dt * 16 + l15;   // dh col; k = key, contiguous in Vt row
            v16bf bfr = as_frag(*(const U128*)(Vt + n * VSTR + 8 * lhalf),
                                *(const U128*)(Vt + n * VSTR + 16 + 8 * lhalf));
            acc[dt] = wmma_bf16(pf, bfr, acc[dt]);
        }
    }

    // normalize and emit bf16 [B*S][D]
#pragma unroll
    for (int j = 0; j < 8; ++j) {
        const float inv = 1.f / (lrow[j] + 1e-20f);
        const int row = qb + wid * 16 + j + 8 * lhalf;
#pragma unroll
        for (int dt = 0; dt < 4; ++dt) {
            const int col = h * DHH + dt * 16 + l15;
            out[(size_t)(b * SS + row) * DD + col] = f2bf(acc[dt][j] * inv);
        }
    }
}

// =====================================================================
// Host-side orchestration
// =====================================================================
extern "C" void kernel_launch(void* const* d_in, const int* in_sizes, int n_in,
                              void* d_out, int out_size, void* d_ws, size_t ws_size,
                              hipStream_t stream) {
    const float* x     = (const float*)d_in[0];
    const int*   mask  = (const int*)  d_in[1];
    const float* w_qkv = (const float*)d_in[2];
    const float* w_out = (const float*)d_in[3];
    const float* b_out = (const float*)d_in[4];
    const float* g1    = (const float*)d_in[5];
    const float* be1   = (const float*)d_in[6];
    const float* g2    = (const float*)d_in[7];
    const float* be2   = (const float*)d_in[8];
    const float* w1    = (const float*)d_in[9];
    const float* b1    = (const float*)d_in[10];
    const float* w2    = (const float*)d_in[11];
    const float* b2    = (const float*)d_in[12];
    float* out = (float*)d_out;

    char* p = (char*)d_ws;
    auto alloc = [&](size_t bytes) -> void* {
        void* r = (void*)p;
        p += (bytes + 255) & ~(size_t)255;
        return r;
    };
    u16*   wqkvT = (u16*)  alloc((size_t)TD * DD * 2);      // [3072][1024]
    u16*   woutT = (u16*)  alloc((size_t)DD * DD * 2);      // [1024][1024]
    u16*   w1T   = (u16*)  alloc((size_t)FFNN * DD * 2);    // [4096][1024]
    u16*   w2T   = (u16*)  alloc((size_t)DD * FFNN * 2);    // [1024][4096]
    u16*   hbuf  = (u16*)  alloc((size_t)MM * DD * 2);      // LN out (reused for LN2)
    u16*   qkvb  = (u16*)  alloc((size_t)MM * TD * 2);      // QKV bf16
    u16*   attnb = (u16*)  alloc((size_t)MM * DD * 2);      // attention out bf16
    float* x1    = (float*)alloc((size_t)MM * DD * 4);      // post-attn residual (fp32)
    u16*   a1    = (u16*)  alloc((size_t)MM * FFNN * 2);    // relu(ffn1) bf16

    // 1) weight convert + transpose to [N][K] bf16
    k_transpose_bf16<<<dim3(TD / 32,   DD / 32),   256, 0, stream>>>(w_qkv, wqkvT, DD, TD);
    k_transpose_bf16<<<dim3(DD / 32,   DD / 32),   256, 0, stream>>>(w_out, woutT, DD, DD);
    k_transpose_bf16<<<dim3(FFNN / 32, DD / 32),   256, 0, stream>>>(w1,    w1T,   DD, FFNN);
    k_transpose_bf16<<<dim3(DD / 32,   FFNN / 32), 256, 0, stream>>>(w2,    w2T,   FFNN, DD);

    // 2) LN1
    k_layernorm_bf16<<<MM, 256, 0, stream>>>(x, g1, be1, hbuf, DD);

    // 3) QKV = h @ Wqkv   (bf16 out, no bias)
    k_gemm_bf16<false, false, false, true><<<dim3(TD / 128, MM / 128), 128, 0, stream>>>(
        hbuf, wqkvT, nullptr, nullptr, (void*)qkvb, MM, TD, DD);

    // 4) attention
    k_attention<<<dim3(SS / 128, HH, BB), 256, 0, stream>>>(qkvb, mask, attnb);

    // 5) x1 = x + attn @ Wout + b_out   (fp32 out)
    k_gemm_bf16<true, false, true, false><<<dim3(DD / 128, MM / 128), 128, 0, stream>>>(
        attnb, woutT, b_out, x, (void*)x1, MM, DD, DD);

    // 6) LN2
    k_layernorm_bf16<<<MM, 256, 0, stream>>>(x1, g2, be2, hbuf, DD);

    // 7) a1 = relu(h2 @ W1 + b1)   (bf16 out)
    k_gemm_bf16<true, true, false, true><<<dim3(FFNN / 128, MM / 128), 128, 0, stream>>>(
        hbuf, w1T, b1, nullptr, (void*)a1, MM, FFNN, DD);

    // 8) out = x1 + a1 @ W2 + b2   (fp32 out)
    k_gemm_bf16<true, false, true, false><<<dim3(DD / 128, MM / 128), 128, 0, stream>>>(
        a1, w2T, b2, x1, (void*)out, MM, DD, FFNN);

    (void)in_sizes; (void)n_in; (void)out_size; (void)ws_size;
}
